// ModulatedConv2d_42064909697592
// MI455X (gfx1250) — compile-verified, matrix operand order checked by reading, not run on hardware
//
#include <hip/hip_runtime.h>
#include <hip/hip_bf16.h>
#include <math.h>

// ---------------- problem constants ----------------
#define BATCH   16
#define IC      256
#define OC      256
#define HH      128
#define KS      3
#define WDIM    512

#define CONV_SCALE (0.02083333333333333333f)   // 1/sqrt(256*9)
#define MOD_SCALE  (0.04419417382415922f)      // 1/sqrt(512)

// ---------------- WMMA types ----------------
typedef __attribute__((ext_vector_type(16))) __bf16 v16bf;
typedef __attribute__((ext_vector_type(8)))  float  v8f;
typedef __attribute__((ext_vector_type(4)))  unsigned int v4u;

union FragBF {
    v16bf h;
    v4u   u[2];
};

struct TapFrags {
    FragBF a[4];   // 4 x (16oc x 32ic)
    FragBF b[2];   // 2 x (32ic x 16px)
};

// ---------------- kernel 1: style = w @ (mod_weight*scale)^T + bias ----------------
__global__ __launch_bounds__(256)
void style_kernel(const float* __restrict__ w, const float* __restrict__ mw,
                  const float* __restrict__ mb, float* __restrict__ style) {
    int b = blockIdx.x;        // 0..15
    int c = threadIdx.x;       // 0..255
    const float* wr = w  + (size_t)b * WDIM;
    const float* mr = mw + (size_t)c * WDIM;
    float s = 0.f;
#pragma unroll 8
    for (int j = 0; j < WDIM; ++j) s += wr[j] * mr[j];
    style[b * IC + c] = s * MOD_SCALE + mb[c];
}

// ---------------- kernel 2: modulate + demodulate, emit bf16 A in [b][tap][oc][ic] ----------------
__global__ __launch_bounds__(256)
void modw_kernel(const float* __restrict__ weight, const float* __restrict__ style,
                 __bf16* __restrict__ Abuf) {
    int oc = blockIdx.x;       // 0..255
    int b  = blockIdx.y;       // 0..15
    int ic = threadIdx.x;      // 0..255

    float s = style[b * IC + ic];
    const float* wp = weight + ((size_t)oc * IC + ic) * (KS * KS);

    float v[9];
    float sum = 0.f;
#pragma unroll
    for (int j = 0; j < 9; ++j) {
        v[j] = CONV_SCALE * wp[j] * s;
        sum += v[j] * v[j];
    }

    __shared__ float red[256];
    red[ic] = sum;
    __syncthreads();
#pragma unroll
    for (int o = 128; o > 0; o >>= 1) {
        if (ic < o) red[ic] += red[ic + o];
        __syncthreads();
    }
    float d = rsqrtf(red[0] + 1e-8f);

#pragma unroll
    for (int j = 0; j < 9; ++j) {
        Abuf[(((size_t)b * 9 + j) * OC + oc) * IC + ic] = (__bf16)(v[j] * d);
    }
}

// ---------------- kernel 3: implicit-GEMM conv via v_wmma_f32_16x16x32_bf16 ----------------
// Block: 256 threads = 8 waves (wave32). C tile = 128 oc x 128 px (one image row).
// Wave (wm in 0..1, wn in 0..3) owns 64 oc x 32 px  ->  4x2 accumulators of 16x16 f32.
// Tap loop is software-pipelined: fragments for tap t+1 load under tap t's WMMAs.
#define IC_STEP 32
#define ICP     40            // padded ic stride (halves); 80B, multiple of 16B
#define SLAB_W  130           // HH + 2 halo columns

__global__ __launch_bounds__(256)
void conv_kernel(const float* __restrict__ x, const __bf16* __restrict__ Abuf,
                 float* __restrict__ out) {
    __shared__ __bf16 slab[3 * SLAB_W * ICP];   // 31,200 bytes

    const int tid  = threadIdx.x;
    const int lane = tid & 31;
    const int wave = tid >> 5;
    const int wm   = wave >> 2;          // 0..1  (M direction)
    const int wn   = wave & 3;           // 0..3  (N direction)
    const int lr   = lane & 15;          // lane % 16
    const int half = lane >> 4;          // 0/1
    const int koff = half * 8;           // K packing split per ISA A/B 16-bit layout

    const int y  = blockIdx.x;           // image row, 0..127
    const int mt = blockIdx.y;           // oc macro-tile, 0..1
    const int b  = blockIdx.z;           // batch

    v8f acc[4][2] = {};

    const float* xb = x + (size_t)b * IC * HH * HH;
    const __bf16* Ab0 = Abuf + (size_t)b * 9 * OC * IC;

    for (int icb = 0; icb < IC / IC_STEP; ++icb) {
        __syncthreads();   // protect slab against previous iteration's readers

        // ---- stage x slab: [yy(3)][px(130)][ic(32)] bf16, halo + zero padding ----
        const int total = 3 * IC_STEP * SLAB_W;            // 12480
        for (int e = tid; e < total; e += 256) {
            int px = e % SLAB_W;
            int t  = e / SLAB_W;
            int ic = t & (IC_STEP - 1);
            int yy = t >> 5;
            int gy = y + yy - 1;
            int gx = px - 1;
            float v = 0.f;
            if ((unsigned)gy < (unsigned)HH && (unsigned)gx < (unsigned)HH)
                v = xb[((size_t)(icb * IC_STEP + ic) * HH + gy) * HH + gx];
            slab[(yy * SLAB_W + px) * ICP + ic] = (__bf16)v;
        }
        __syncthreads();

        // ---- prefetch next ic-block's x lines toward the WGP (global_prefetch) ----
        if (icb + 1 < IC / IC_STEP) {
            const float* nb = xb + (size_t)(icb + 1) * IC_STEP * HH * HH;
#pragma unroll
            for (int i = 0; i < 3; ++i) {
                int e  = tid + i * 4160;            // spread across the slab footprint
                int px = e % SLAB_W;
                int t2 = e / SLAB_W;
                int ic = t2 & (IC_STEP - 1);
                int yy = t2 >> 5;
                int gy = y + yy - 1;
                int gx = px - 1;
                if ((unsigned)gy < (unsigned)HH && (unsigned)gx < (unsigned)HH)
                    __builtin_prefetch(&nb[((size_t)ic * HH + gy) * HH + gx], 0, 3);
            }
        }

        // ---- per-icb invariant fragment base pointers ----
        const __bf16* pA[4];
#pragma unroll
        for (int m16 = 0; m16 < 4; ++m16) {
            int oc = mt * 128 + wm * 64 + m16 * 16 + lr;
            pA[m16] = Ab0 + (size_t)oc * IC + icb * IC_STEP + koff;
        }
        const __bf16* pB[2];
#pragma unroll
        for (int n16 = 0; n16 < 2; ++n16) {
            int xc = wn * 32 + n16 * 16 + lr;       // output column
            pB[n16] = &slab[xc * ICP + koff];
        }

        // ---- software-pipelined 9-tap loop; each tap = one K=32 WMMA step ----
        TapFrags cur, nxt;
        {
#pragma unroll
            for (int m16 = 0; m16 < 4; ++m16) {
                const __bf16* p = pA[m16];                       // tap 0
                cur.a[m16].u[0] = *(const v4u*)(p);
                cur.a[m16].u[1] = *(const v4u*)(p + 16);
            }
#pragma unroll
            for (int n16 = 0; n16 < 2; ++n16) {
                const __bf16* p = pB[n16];                       // ky=0,kx=0
                cur.b[n16].u[0] = *(const v4u*)(p);
                cur.b[n16].u[1] = *(const v4u*)(p + 16);
            }
        }

#pragma unroll
        for (int t = 0; t < 9; ++t) {
            if (t < 8) {
                const int tn = t + 1;
                const int ky = tn / 3, kx = tn % 3;
                const size_t aoff = (size_t)tn * OC * IC;
                const int    boff = (ky * SLAB_W + kx) * ICP;
#pragma unroll
                for (int m16 = 0; m16 < 4; ++m16) {
                    const __bf16* p = pA[m16] + aoff;
                    nxt.a[m16].u[0] = *(const v4u*)(p);
                    nxt.a[m16].u[1] = *(const v4u*)(p + 16);
                }
#pragma unroll
                for (int n16 = 0; n16 < 2; ++n16) {
                    const __bf16* p = pB[n16] + boff;
                    nxt.b[n16].u[0] = *(const v4u*)(p);
                    nxt.b[n16].u[1] = *(const v4u*)(p + 16);
                }
            }

#pragma unroll
            for (int m16 = 0; m16 < 4; ++m16)
#pragma unroll
                for (int n16 = 0; n16 < 2; ++n16)
                    acc[m16][n16] = __builtin_amdgcn_wmma_f32_16x16x32_bf16(
                        false, cur.a[m16].h, false, cur.b[n16].h,
                        (short)0, acc[m16][n16], false, false);

            if (t < 8) cur = nxt;
        }
    }

    // ---- store C per ISA f32 C/D layout: VGPR r -> M = r + 8*half, N = lr ----
    float* ob = out + (size_t)b * OC * HH * HH;
#pragma unroll
    for (int m16 = 0; m16 < 4; ++m16) {
#pragma unroll
        for (int n16 = 0; n16 < 2; ++n16) {
            int xc = wn * 32 + n16 * 16 + lr;
#pragma unroll
            for (int r = 0; r < 8; ++r) {
                int oc = mt * 128 + wm * 64 + m16 * 16 + r + half * 8;
                ob[((size_t)oc * HH + y) * HH + xc] = acc[m16][n16][r];
            }
        }
    }
}

// ---------------- launcher ----------------
extern "C" void kernel_launch(void* const* d_in, const int* in_sizes, int n_in,
                              void* d_out, int out_size, void* d_ws, size_t ws_size,
                              hipStream_t stream) {
    const float* x   = (const float*)d_in[0];   // [16,256,128,128]
    const float* w   = (const float*)d_in[1];   // [16,512]
    const float* wt  = (const float*)d_in[2];   // [1,256,256,3,3]
    const float* mw  = (const float*)d_in[3];   // [256,512]
    const float* mb  = (const float*)d_in[4];   // [256]
    float* out = (float*)d_out;                 // [16,256,128,128]

    float*  style = (float*)d_ws;                              // 16*256*4 = 16 KB
    __bf16* Abuf  = (__bf16*)((char*)d_ws + 16 * 1024);        // 16*9*256*256*2 ≈ 18.9 MB

    style_kernel<<<dim3(BATCH), dim3(256), 0, stream>>>(w, mw, mb, style);
    modw_kernel<<<dim3(OC, BATCH), dim3(256), 0, stream>>>(wt, style, Abuf);
    conv_kernel<<<dim3(HH, 2, BATCH), dim3(256), 0, stream>>>(x, Abuf, out);
}
